// EruSelfAttentionModel_87909390614745
// MI455X (gfx1250) — matrix operand order repaired and checked by reading, not compiled, not run on hardware
//
#include <hip/hip_runtime.h>
#include <hip/hip_bf16.h>

// ---------------------------------------------------------------------------
// Problem constants (match reference)
// ---------------------------------------------------------------------------
constexpr int Bc = 4;      // batch
constexpr int Sc = 1024;   // sequence
constexpr int Ec = 1024;   // embedding
constexpr int Ac = 64;     // attention dim
constexpr int Hc = 16;     // heads
constexpr int Lc = 2;      // layers
constexpr float EPSc = 1e-5f;

typedef __attribute__((ext_vector_type(16))) __bf16 v16bf;
typedef __attribute__((ext_vector_type(8)))  float  v8f;

union Frag16 { v16bf v; unsigned u[8]; };

__device__ __forceinline__ unsigned short f32_to_bf16(float f) {
    union { float f; unsigned u; } v; v.f = f;
    unsigned rounding = 0x7FFFu + ((v.u >> 16) & 1u);   // round-to-nearest-even
    return (unsigned short)((v.u + rounding) >> 16);
}

// ---------------------------------------------------------------------------
// fp32 -> bf16 cast (weights)
// ---------------------------------------------------------------------------
__global__ __launch_bounds__(256)
void cast_bf16_kernel(const float* __restrict__ in, unsigned short* __restrict__ out, long n) {
    long i = (long)blockIdx.x * 256 + threadIdx.x;
    if (i < n) out[i] = f32_to_bf16(in[i]);
}

// ---------------------------------------------------------------------------
// Embedding gather, replicated across heads: h[b,h,s,:] = emb[x[b,s],:]
// grid.x = B*S
// ---------------------------------------------------------------------------
__global__ __launch_bounds__(256)
void embed_kernel(const int* __restrict__ x, const float* __restrict__ emb,
                  float* __restrict__ h) {
    int row = blockIdx.x;                 // b*S + s
    int b = row >> 10, s = row & 1023;
    int tok = x[row];
    const float* e = emb + (size_t)tok * Ec;
    for (int j = threadIdx.x; j < Ec; j += 256) {
        float val = e[j];
#pragma unroll
        for (int hh = 0; hh < Hc; ++hh)
            h[((size_t)(b * Hc + hh) * Sc + s) * Ec + j] = val;
    }
}

// ---------------------------------------------------------------------------
// LayerNorm over E, one wave32 per row, output bf16.
// grid.x = (B*H*S)/8, block = 256 (8 waves -> 8 rows)
// ---------------------------------------------------------------------------
__global__ __launch_bounds__(256)
void layernorm_kernel(const float* __restrict__ h, const float* __restrict__ gamma,
                      const float* __restrict__ beta, unsigned short* __restrict__ hn) {
    int row  = blockIdx.x * 8 + (threadIdx.x >> 5);
    int lane = threadIdx.x & 31;
    const float* r = h + (size_t)row * Ec;
    float s = 0.f, ss = 0.f;
    for (int e = lane; e < Ec; e += 32) { float v = r[e]; s += v; ss += v * v; }
#pragma unroll
    for (int o = 16; o; o >>= 1) { s += __shfl_xor(s, o, 32); ss += __shfl_xor(ss, o, 32); }
    float mu  = s  * (1.f / Ec);
    float var = ss * (1.f / Ec) - mu * mu;
    float rs  = rsqrtf(var + EPSc);
    unsigned short* dst = hn + (size_t)row * Ec;
    for (int e = lane; e < Ec; e += 32) {
        float v = (r[e] - mu) * rs * gamma[e] + beta[e];
        dst[e] = f32_to_bf16(v);
    }
}

// ---------------------------------------------------------------------------
// Row softmax over 1024 fp32 scores -> bf16 probs. grid.x = B*H*S, block=256.
// ---------------------------------------------------------------------------
__global__ __launch_bounds__(256)
void softmax_kernel(const float* __restrict__ S, unsigned short* __restrict__ P) {
    __shared__ float red[8];
    int tid = threadIdx.x;
    const float* r = S + (size_t)blockIdx.x * Sc;
    float v[4];
#pragma unroll
    for (int i = 0; i < 4; ++i) v[i] = r[tid + i * 256];
    float m = fmaxf(fmaxf(v[0], v[1]), fmaxf(v[2], v[3]));
#pragma unroll
    for (int o = 16; o; o >>= 1) m = fmaxf(m, __shfl_xor(m, o, 32));
    if ((tid & 31) == 0) red[tid >> 5] = m;
    __syncthreads();
    m = red[0];
#pragma unroll
    for (int w = 1; w < 8; ++w) m = fmaxf(m, red[w]);
    float sum = 0.f;
#pragma unroll
    for (int i = 0; i < 4; ++i) { v[i] = __expf(v[i] - m); sum += v[i]; }
#pragma unroll
    for (int o = 16; o; o >>= 1) sum += __shfl_xor(sum, o, 32);
    __syncthreads();
    if ((tid & 31) == 0) red[tid >> 5] = sum;
    __syncthreads();
    sum = red[0];
#pragma unroll
    for (int w = 1; w < 8; ++w) sum += red[w];
    float inv = 1.f / sum;
    unsigned short* p = P + (size_t)blockIdx.x * Sc;
#pragma unroll
    for (int i = 0; i < 4; ++i) p[tid + i * 256] = f32_to_bf16(v[i] * inv);
}

// ---------------------------------------------------------------------------
// bf16 WMMA GEMM: C[M,N] = alpha * A[M,K] * B    (row-major A, ld = lda)
//   BT=true : B given as W[N,K] row-major (ld = ldb)   -> C = A * W^T
//   BT=false: B given as V[K,N] row-major (ld = ldb)   -> C = A * V
// 128x64 C tile per block (256 threads = 8 wave32, 4x2 grid of 32x32 wave
// tiles -> 4 accumulators/wave, 2 A-frags x 2 B-frags reused), K-step 32.
// z-batch: A += z*aPerZ; B += z*bPerZ + (z%Hdiv)*bPerH; C += z*cPerZ.
// ---------------------------------------------------------------------------
constexpr int LDT = 40;  // padded LDS row stride (ushorts), even for dword reads

template <bool BT, bool OUT_BF16>
__global__ __launch_bounds__(256)
void wmma_gemm_kernel(const unsigned short* __restrict__ Ag, int lda, long aPerZ,
                      const unsigned short* __restrict__ Bg, int ldb, long bPerZ, long bPerH,
                      void* __restrict__ Cg, int ldc, long cPerZ,
                      int K, int Hdiv, float alpha) {
    __shared__ unsigned short As[128][LDT];
    __shared__ unsigned short Bs[64][LDT];

    const int z   = blockIdx.z;
    const int hh  = z % Hdiv;
    const unsigned short* A  = Ag + (size_t)z * aPerZ;
    const unsigned short* Bp = Bg + (size_t)z * bPerZ + (size_t)hh * bPerH;

    const int m0g = blockIdx.y * 128;
    const int n0g = blockIdx.x * 64;

    const int tid  = threadIdx.x;
    const int wave = tid >> 5;
    const int lane = tid & 31;
    const int wm   = (wave & 3) * 32;   // wave tile M offset within block tile
    const int wn   = (wave >> 2) * 32;  // wave tile N offset within block tile
    const int rl   = lane & 15;         // row within 16x16 for A / col for B,C
    const int hi   = lane >> 4;         // lane-half selector

    v8f acc00 = {}, acc01 = {}, acc10 = {}, acc11 = {};

    for (int k0 = 0; k0 < K; k0 += 32) {
        __syncthreads();
        // ---- stage A tile (128 rows x 32 K): 2x 16B chunks per thread ----
        {
#pragma unroll
            for (int cchunk = 0; cchunk < 2; ++cchunk) {
                int c   = tid + cchunk * 256;       // 512 chunks of 8 ushorts
                int row = c >> 2, col = (c & 3) * 8;
                const uint4* src = reinterpret_cast<const uint4*>(
                    A + (size_t)(m0g + row) * lda + k0 + col);
                *reinterpret_cast<uint4*>(&As[row][col]) = *src;
                if (k0 + 32 < K)
                    __builtin_prefetch(A + (size_t)(m0g + row) * lda + k0 + 32 + col, 0, 3);
            }
        }
        // ---- stage B tile as Bs[n][k] (64 x 32) ----
        if (BT) {
            int r = tid >> 2, c = (tid & 3) * 8;
            const uint4* src = reinterpret_cast<const uint4*>(
                Bp + (size_t)(n0g + r) * ldb + k0 + c);
            *reinterpret_cast<uint4*>(&Bs[r][c]) = *src;
        } else {
            int kk = tid >> 3, nb = (tid & 7) * 8;
            union { uint4 q; unsigned short s[8]; } tmp;
            tmp.q = *reinterpret_cast<const uint4*>(
                Bp + (size_t)(k0 + kk) * ldb + n0g + nb);
#pragma unroll
            for (int i = 0; i < 8; ++i) Bs[nb + i][kk] = tmp.s[i];
        }
        __syncthreads();

        // ---- fragments: 2 A (16x32) + 2 B (32x16), documented wave32 layout ----
        Frag16 af0, af1, bf0, bf1;
        {
            const unsigned* Arow0 = reinterpret_cast<const unsigned*>(&As[wm + rl][0]);
            const unsigned* Arow1 = reinterpret_cast<const unsigned*>(&As[wm + 16 + rl][0]);
#pragma unroll
            for (int j = 0; j < 4; ++j) {
                af0.u[j]     = Arow0[hi * 4 + j];       // K = hi*8 + {0..7}
                af0.u[4 + j] = Arow0[8 + hi * 4 + j];   // K = 16 + hi*8 + {0..7}
                af1.u[j]     = Arow1[hi * 4 + j];
                af1.u[4 + j] = Arow1[8 + hi * 4 + j];
            }
            const unsigned* Brow0 = reinterpret_cast<const unsigned*>(&Bs[wn + rl][0]);
            const unsigned* Brow1 = reinterpret_cast<const unsigned*>(&Bs[wn + 16 + rl][0]);
#pragma unroll
            for (int j = 0; j < 4; ++j) {
                bf0.u[j]     = Brow0[hi * 4 + j];       // K = hi*8 + {0..7}
                bf0.u[4 + j] = Brow0[8 + hi * 4 + j];   // K = 16 + hi*8 + {0..7}
                bf1.u[j]     = Brow1[hi * 4 + j];
                bf1.u[4 + j] = Brow1[8 + hi * 4 + j];
            }
        }
        acc00 = __builtin_amdgcn_wmma_f32_16x16x32_bf16(
            false, af0.v, false, bf0.v, (short)0, acc00, false, false);
        acc01 = __builtin_amdgcn_wmma_f32_16x16x32_bf16(
            false, af0.v, false, bf1.v, (short)0, acc01, false, false);
        acc10 = __builtin_amdgcn_wmma_f32_16x16x32_bf16(
            false, af1.v, false, bf0.v, (short)0, acc10, false, false);
        acc11 = __builtin_amdgcn_wmma_f32_16x16x32_bf16(
            false, af1.v, false, bf1.v, (short)0, acc11, false, false);
    }

    // ---- epilogue: C/D layout — VGPR j holds row hi*8+j, col = lane&15 ----
    const int r0 = m0g + wm + hi * 8;        // M-subtile 0 row base (per-lane)
    const int r1 = r0 + 16;                  // M-subtile 1
    const int c0 = n0g + wn + rl;            // N-subtile 0 col
    const int c1 = c0 + 16;                  // N-subtile 1
    if (OUT_BF16) {
        unsigned short* C = (unsigned short*)Cg + (size_t)z * cPerZ;
#pragma unroll
        for (int j = 0; j < 8; ++j) {
            C[(size_t)(r0 + j) * ldc + c0] = f32_to_bf16(acc00[j] * alpha);
            C[(size_t)(r0 + j) * ldc + c1] = f32_to_bf16(acc01[j] * alpha);
            C[(size_t)(r1 + j) * ldc + c0] = f32_to_bf16(acc10[j] * alpha);
            C[(size_t)(r1 + j) * ldc + c1] = f32_to_bf16(acc11[j] * alpha);
        }
    } else {
        float* C = (float*)Cg + (size_t)z * cPerZ;
#pragma unroll
        for (int j = 0; j < 8; ++j) {
            C[(size_t)(r0 + j) * ldc + c0] = acc00[j] * alpha;
            C[(size_t)(r0 + j) * ldc + c1] = acc01[j] * alpha;
            C[(size_t)(r1 + j) * ldc + c0] = acc10[j] * alpha;
            C[(size_t)(r1 + j) * ldc + c1] = acc11[j] * alpha;
        }
    }
}

// ---------------------------------------------------------------------------
// Host-side orchestration
// ---------------------------------------------------------------------------
extern "C" void kernel_launch(void* const* d_in, const int* in_sizes, int n_in,
                              void* d_out, int out_size, void* d_ws, size_t ws_size,
                              hipStream_t stream) {
    (void)in_sizes; (void)n_in; (void)out_size; (void)ws_size;

    const int*   x        = (const int*)  d_in[0];
    const float* emb      = (const float*)d_in[1];
    const float* ln_gamma = (const float*)d_in[2];
    const float* ln_beta  = (const float*)d_in[3];
    const float* Wq       = (const float*)d_in[4];
    const float* Wk       = (const float*)d_in[5];
    const float* Wv       = (const float*)d_in[6];
    float*       out      = (float*)d_out;

    const size_t BHSE = (size_t)Bc * Hc * Sc * Ec;     // 67,108,864
    const size_t BHSA = (size_t)Bc * Hc * Sc * Ac;
    const size_t BHSS = (size_t)Bc * Hc * Sc * Sc;
    const long   nWq  = (long)Lc * Hc * Ac * Ec;
    const long   nWv  = (long)Lc * Hc * Ec * Ec;

    char* ws = (char*)d_ws;
    auto alloc = [&](size_t bytes) -> char* {
        char* p = ws;
        ws += (bytes + 255) & ~(size_t)255;
        return p;
    };

    float*          h0     = (float*)         alloc(BHSE * 4);
    float*          h1     = (float*)         alloc(BHSE * 4);
    unsigned short* hn     = (unsigned short*)alloc(BHSE * 2);
    unsigned short* qb     = (unsigned short*)alloc(BHSA * 2);
    unsigned short* kb     = (unsigned short*)alloc(BHSA * 2);
    unsigned short* vb     = (unsigned short*)alloc(BHSE * 2);
    float*          scores = (float*)         alloc(BHSS * 4);
    unsigned short* probs  = (unsigned short*)alloc(BHSS * 2);
    unsigned short* wq16   = (unsigned short*)alloc((size_t)nWq * 2);
    unsigned short* wk16   = (unsigned short*)alloc((size_t)nWq * 2);
    unsigned short* wv16   = (unsigned short*)alloc((size_t)nWv * 2);

    // weights fp32 -> bf16
    cast_bf16_kernel<<<(nWq + 255) / 256, 256, 0, stream>>>(Wq, wq16, nWq);
    cast_bf16_kernel<<<(nWq + 255) / 256, 256, 0, stream>>>(Wk, wk16, nWq);
    cast_bf16_kernel<<<(nWv + 255) / 256, 256, 0, stream>>>(Wv, wv16, nWv);

    // embedding gather replicated across heads
    embed_kernel<<<Bc * Sc, 256, 0, stream>>>(x, emb, h0);

    const float inv_scale = 1.0f / 32.0f;  // 1/sqrt(E)
    float* cur = h0;

    for (int l = 0; l < Lc; ++l) {
        // LayerNorm -> bf16
        layernorm_kernel<<<(Bc * Hc * Sc) / 8, 256, 0, stream>>>(
            cur, ln_gamma + (size_t)l * Ec, ln_beta + (size_t)l * Ec, hn);

        const unsigned short* wql = wq16 + (size_t)l * Hc * Ac * Ec;
        const unsigned short* wkl = wk16 + (size_t)l * Hc * Ac * Ec;
        const unsigned short* wvl = wv16 + (size_t)l * Hc * Ec * Ec;

        // Q = hn * Wq^T : [S,E] x [A,E]^T -> [S,A] bf16
        wmma_gemm_kernel<true, true><<<dim3(Ac / 64, Sc / 128, Bc * Hc), 256, 0, stream>>>(
            hn, Ec, (long)Sc * Ec,
            wql, Ec, 0, (long)Ac * Ec,
            qb, Ac, (long)Sc * Ac,
            Ec, Hc, 1.0f);
        // K = hn * Wk^T
        wmma_gemm_kernel<true, true><<<dim3(Ac / 64, Sc / 128, Bc * Hc), 256, 0, stream>>>(
            hn, Ec, (long)Sc * Ec,
            wkl, Ec, 0, (long)Ac * Ec,
            kb, Ac, (long)Sc * Ac,
            Ec, Hc, 1.0f);
        // V = hn * Wv^T : [S,E] x [E,E]^T -> [S,E] bf16
        wmma_gemm_kernel<true, true><<<dim3(Ec / 64, Sc / 128, Bc * Hc), 256, 0, stream>>>(
            hn, Ec, (long)Sc * Ec,
            wvl, Ec, 0, (long)Ec * Ec,
            vb, Ec, (long)Sc * Ec,
            Ec, Hc, 1.0f);

        // scores = (Q * K^T) / sqrt(E) : [S,A] x [S,A]^T -> [S,S] fp32
        wmma_gemm_kernel<true, false><<<dim3(Sc / 64, Sc / 128, Bc * Hc), 256, 0, stream>>>(
            qb, Ac, (long)Sc * Ac,
            kb, Ac, (long)Sc * Ac, 0,
            scores, Sc, (long)Sc * Sc,
            Ac, Hc, inv_scale);

        // softmax rows -> bf16 probs
        softmax_kernel<<<Bc * Hc * Sc, 256, 0, stream>>>(scores, probs);

        // h_next = probs * V : [S,S] x [S,E] -> [S,E] fp32
        float* nxt = (l == Lc - 1) ? out : ((cur == h0) ? h1 : h0);
        wmma_gemm_kernel<false, false><<<dim3(Ec / 64, Sc / 128, Bc * Hc), 256, 0, stream>>>(
            probs, Sc, (long)Sc * Sc,
            vb, Ec, (long)Sc * Ec, 0,
            nxt, Ec, (long)Sc * Ec,
            Sc, Hc, 1.0f);
        cur = nxt;
    }
}
